// VAE_37915971289461
// MI455X (gfx1250) — matrix operand order repaired and checked by reading, not compiled, not run on hardware
//
#include <hip/hip_runtime.h>
#include <math.h>

// ---------------- types ----------------
typedef __attribute__((ext_vector_type(16))) __bf16 bf16x16;
typedef __attribute__((ext_vector_type(8)))  __bf16 bf16x8;
typedef __attribute__((ext_vector_type(8)))  float  f32x8;

#define NB 32   // batch
#define NT 64   // time
#define NIN 90
#define NU 4
#define NSEG 16
#define NH1 2048
#define NHC 1024
#define NHD 1024
#define NZ 512

// ---------------- WMMA helpers ----------------
static __device__ inline f32x8 wmma_bf16(bf16x16 a, bf16x16 b, f32x8 c) {
  // D = A(16x32 bf16) x B(32x16 bf16) + C(16x16 f32)
  return __builtin_amdgcn_wmma_f32_16x16x32_bf16(
      false, a, false, b, (short)0, c, false, false);
}

// A-fragment: row-major bf16 activations [rows, ld]; lane l holds M=l&15,
// K halves selected by l>>4 (comps 0-7 -> k0+koff.., comps 8-15 -> k0+16+koff..)
static __device__ inline bf16x16 load_a(const __bf16* __restrict__ act, int ld,
                                        int row, int k0, int koff) {
  const bf16x8 lo = *reinterpret_cast<const bf16x8*>(act + (size_t)row * ld + k0 + koff);
  const bf16x8 hi = *reinterpret_cast<const bf16x8*>(act + (size_t)row * ld + k0 + 16 + koff);
  bf16x16 r;
#pragma unroll
  for (int i = 0; i < 8; ++i) { r[i] = lo[i]; r[i + 8] = hi[i]; }
  return r;
}

// B-fragment: pre-packed weights, 512 bf16 per (ntile,kblock), [lane][comp] contiguous
static __device__ inline bf16x16 load_b(const __bf16* __restrict__ Wp, int nkb,
                                        int nt, int kb, int lid) {
  return *reinterpret_cast<const bf16x16*>(
      Wp + ((((size_t)nt * nkb + kb) << 9) + (size_t)lid * 16));
}

static __device__ inline float sigm(float x) { return 1.f / (1.f + expf(-x)); }

// ---------------- weight packer: fp32 [N,K] -> bf16 WMMA-B fragments ----------------
__global__ void pack_w_kernel(const float* __restrict__ W, __bf16* __restrict__ out,
                              int N, int K, int nkb, size_t total) {
  size_t idx = (size_t)blockIdx.x * blockDim.x + threadIdx.x;
  if (idx >= total) return;
  size_t blk = idx >> 9;
  int e = (int)(idx & 511);
  int kb = (int)(blk % nkb);
  int nt = (int)(blk / nkb);
  int lane = e >> 4, c = e & 15;
  int nn = nt * 16 + (lane & 15), kk = kb * 32 + ((c >> 3) << 4) + ((lane >> 4) << 3) + (c & 7);
  float v = (nn < N && kk < K) ? W[(size_t)nn * K + kk] : 0.f;
  out[idx] = (__bf16)v;
}

// ---------------- fused LSTM step ----------------
// gates = bih+bhh + xin@Wih^T + h@Whh^T ; c' = s(f)c + s(i)tanh(g) ; h' = s(o)tanh(c')
// 4 waves/block, 1 wave = 16 hidden columns (all 4 gates), grid.x = H/64
__global__ __launch_bounds__(128) void lstm_step_kernel(
    const __bf16* __restrict__ xin, int Kx, const __bf16* __restrict__ Wih,
    const __bf16* __restrict__ hin, int H,  const __bf16* __restrict__ Whh,
    const float* __restrict__ bih, const float* __restrict__ bhh,
    float* __restrict__ c_state, __bf16* __restrict__ hout,
    float* __restrict__ hout_f32, __bf16* __restrict__ ys, int ys_ld) {
  const int lid  = threadIdx.x & 31;
  const int jt   = blockIdx.x * 4 + (threadIdx.x >> 5);
  const int j0   = jt * 16;
  const int ntH  = H >> 4;
  const int n    = lid & 15;
  const int j    = j0 + n;
  const int mloc = lid & 15;
  const int koff = (lid >> 4) * 8;

  f32x8 acc[4][2];
#pragma unroll
  for (int g = 0; g < 4; ++g) {
    float bv = bih[g * H + j] + bhh[g * H + j];
#pragma unroll
    for (int rt = 0; rt < 2; ++rt)
#pragma unroll
      for (int v = 0; v < 8; ++v) acc[g][rt][v] = bv;
  }

  if (Kx > 0) {
    const int nkb = Kx >> 5;
    for (int kb = 0; kb < nkb; ++kb) {
      bf16x16 a0 = load_a(xin, Kx, mloc,      kb << 5, koff);
      bf16x16 a1 = load_a(xin, Kx, 16 + mloc, kb << 5, koff);
#pragma unroll
      for (int g = 0; g < 4; ++g) {
        bf16x16 bf = load_b(Wih, nkb, g * ntH + jt, kb, lid);
        acc[g][0] = wmma_bf16(a0, bf, acc[g][0]);
        acc[g][1] = wmma_bf16(a1, bf, acc[g][1]);
      }
    }
  }
  {
    const int nkb = H >> 5;
    for (int kb = 0; kb < nkb; ++kb) {
      bf16x16 a0 = load_a(hin, H, mloc,      kb << 5, koff);
      bf16x16 a1 = load_a(hin, H, 16 + mloc, kb << 5, koff);
#pragma unroll
      for (int g = 0; g < 4; ++g) {
        bf16x16 bf = load_b(Whh, nkb, g * ntH + jt, kb, lid);
        acc[g][0] = wmma_bf16(a0, bf, acc[g][0]);
        acc[g][1] = wmma_bf16(a1, bf, acc[g][1]);
      }
    }
  }

  const int rbase = ((lid >> 4) & 1) * 8;
#pragma unroll
  for (int rt = 0; rt < 2; ++rt) {
#pragma unroll
    for (int v = 0; v < 8; ++v) {
      int row = rt * 16 + rbase + v;
      float gi = acc[0][rt][v], gf = acc[1][rt][v];
      float gg = acc[2][rt][v], go = acc[3][rt][v];
      float cold = c_state[(size_t)row * H + j];
      float cnew = sigm(gf) * cold + sigm(gi) * tanhf(gg);
      float h    = sigm(go) * tanhf(cnew);
      c_state[(size_t)row * H + j] = cnew;
      hout[(size_t)row * H + j]    = (__bf16)h;
      if (hout_f32) hout_f32[(size_t)row * H + j] = h;
      if (ys) ys[(size_t)row * ys_ld + j] = (__bf16)h;
    }
  }
}

// ---------------- generic GEMM: out = act(A[32,K] @ W^T + b), W packed ----------------
__global__ __launch_bounds__(128) void gemm_kernel(
    const __bf16* __restrict__ A, int K, const __bf16* __restrict__ Wp,
    const float* __restrict__ bias, int N, int nbias,
    float* __restrict__ out, int act_tanh) {
  const int lid = threadIdx.x & 31;
  const int jt  = blockIdx.x * 4 + (threadIdx.x >> 5);
  if (jt >= (N >> 4)) return;
  const int n   = lid & 15;
  const int col = jt * 16 + n;
  const int mloc = lid & 15;
  const int koff = (lid >> 4) * 8;

  f32x8 acc[2];
  float bv = (col < nbias) ? bias[col] : 0.f;
#pragma unroll
  for (int rt = 0; rt < 2; ++rt)
#pragma unroll
    for (int v = 0; v < 8; ++v) acc[rt][v] = bv;

  const int nkb = K >> 5;
  for (int kb = 0; kb < nkb; ++kb) {
    bf16x16 a0 = load_a(A, K, mloc,      kb << 5, koff);
    bf16x16 a1 = load_a(A, K, 16 + mloc, kb << 5, koff);
    bf16x16 bf = load_b(Wp, nkb, jt, kb, lid);
    acc[0] = wmma_bf16(a0, bf, acc[0]);
    acc[1] = wmma_bf16(a1, bf, acc[1]);
  }

  const int rbase = ((lid >> 4) & 1) * 8;
#pragma unroll
  for (int rt = 0; rt < 2; ++rt)
#pragma unroll
    for (int v = 0; v < 8; ++v) {
      int row = rt * 16 + rbase + v;
      float x = acc[rt][v];
      if (act_tanh) x = tanhf(x);
      out[(size_t)row * N + col] = x;
    }
}

// ---------------- small elementwise kernels ----------------
__global__ void conv_x_kernel(const float* __restrict__ x, __bf16* __restrict__ dst) {
  int idx = blockIdx.x * blockDim.x + threadIdx.x;           // [T][32][96]
  if (idx >= NT * NB * 96) return;
  int t = idx / (NB * 96), r = idx % (NB * 96), b = r / 96, k = r % 96;
  dst[idx] = (k < NIN) ? (__bf16)x[((size_t)b * NT + t) * NIN + k] : (__bf16)0.0f;
}

__global__ void f32_to_bf16_2d(const float* __restrict__ src, __bf16* __restrict__ dst,
                               int W, int dst_ld, int dst_col, int n) {
  int idx = blockIdx.x * blockDim.x + threadIdx.x;
  if (idx >= n) return;
  int b = idx / W, j = idx % W;
  dst[(size_t)b * dst_ld + dst_col + j] = (__bf16)src[idx];
}

__global__ void z_combine_kernel(const float* __restrict__ zm, const float* __restrict__ s,
                                 const float* __restrict__ eps, float* __restrict__ out_zm,
                                 float* __restrict__ out_zs, __bf16* __restrict__ z_bf) {
  int idx = blockIdx.x * blockDim.x + threadIdx.x;
  if (idx >= NB * NZ) return;
  float m = zm[idx];
  float sp = log1pf(expf(s[idx]));   // softplus
  out_zm[idx] = m;
  out_zs[idx] = sp;
  z_bf[idx] = (__bf16)(m + eps[idx] * sp);
}

// t[:,0:H]->hA, t[:,H:2H]->hB, t[:,2H:3H]->cA, t[:,3H:4H]->cB   (order h1,h2,c1,c2)
__global__ void init_states_kernel(const float* __restrict__ t4,
                                   __bf16* __restrict__ hA, __bf16* __restrict__ hB,
                                   float* __restrict__ cA, float* __restrict__ cB, int H) {
  int idx = blockIdx.x * blockDim.x + threadIdx.x;
  if (idx >= NB * H) return;
  int b = idx / H, j = idx % H;
  const float* row = t4 + (size_t)b * 4 * H;
  hA[idx] = (__bf16)row[j];
  hB[idx] = (__bf16)row[H + j];
  cA[idx] = row[2 * H + j];
  cB[idx] = row[3 * H + j];
}

__global__ void build_xin_kernel(const __bf16* __restrict__ emb_u,
                                 const float* __restrict__ p,
                                 __bf16* __restrict__ xin) {
  int idx = blockIdx.x * blockDim.x + threadIdx.x;         // [32][1120]
  if (idx >= NB * 1120) return;
  int b = idx / 1120, k = idx % 1120;
  __bf16 v;
  if (k < NHC)            v = emb_u[(size_t)b * NHC + k];
  else if (k < NHC + NIN) v = (__bf16)p[b * 96 + (k - NHC)];
  else                    v = (__bf16)0.0f;
  xin[idx] = v;
}

__global__ void copy_p_kernel(const float* __restrict__ p, float* __restrict__ out, int t) {
  int idx = blockIdx.x * blockDim.x + threadIdx.x;
  if (idx >= NB * NIN) return;
  int b = idx / NIN, j = idx % NIN;
  out[((size_t)b * NT + t) * NIN + j] = p[b * 96 + j];
}

__global__ void softmax_kernel(float* __restrict__ out) {
  int idx = blockIdx.x * blockDim.x + threadIdx.x;          // one thread per (b,t)
  if (idx >= NB * NT) return;
  float* row = out + (size_t)idx * NIN;
  float m = row[0];
  for (int j = 1; j < NIN; ++j) m = fmaxf(m, row[j]);
  float sum = 0.f;
  for (int j = 0; j < NIN; ++j) { float e = expf(row[j] - m); row[j] = e; sum += e; }
  float inv = 1.f / sum;
  for (int j = 0; j < NIN; ++j) row[j] *= inv;
}

// ---------------- host ----------------
extern "C" void kernel_launch(void* const* d_in, const int* in_sizes, int n_in,
                              void* d_out, int out_size, void* d_ws, size_t ws_size,
                              hipStream_t stream) {
  (void)in_sizes; (void)n_in; (void)out_size; (void)ws_size;
  const float* x   = (const float*)d_in[0];
  const float* eps = (const float*)d_in[1];
  const float *e0f_wih=(const float*)d_in[2],  *e0f_whh=(const float*)d_in[3],
              *e0f_bih=(const float*)d_in[4],  *e0f_bhh=(const float*)d_in[5];
  const float *e0b_wih=(const float*)d_in[6],  *e0b_whh=(const float*)d_in[7],
              *e0b_bih=(const float*)d_in[8],  *e0b_bhh=(const float*)d_in[9];
  const float *e1f_wih=(const float*)d_in[10], *e1f_whh=(const float*)d_in[11],
              *e1f_bih=(const float*)d_in[12], *e1f_bhh=(const float*)d_in[13];
  const float *e1b_wih=(const float*)d_in[14], *e1b_whh=(const float*)d_in[15],
              *e1b_bih=(const float*)d_in[16], *e1b_bhh=(const float*)d_in[17];
  const float *wm=(const float*)d_in[18], *bm=(const float*)d_in[19];
  const float *ws_=(const float*)d_in[20], *bs=(const float*)d_in[21];
  const float *w2=(const float*)d_in[22], *b2=(const float*)d_in[23];
  const float *c0_whh=(const float*)d_in[25],
              *c0_bih=(const float*)d_in[26], *c0_bhh=(const float*)d_in[27];
  const float *c1_wih=(const float*)d_in[28], *c1_whh=(const float*)d_in[29],
              *c1_bih=(const float*)d_in[30], *c1_bhh=(const float*)d_in[31];
  const float *w3=(const float*)d_in[32], *b3=(const float*)d_in[33];
  const float *d1_wih=(const float*)d_in[34], *d1_whh=(const float*)d_in[35],
              *d1_bih=(const float*)d_in[36], *d1_bhh=(const float*)d_in[37];
  const float *d2_wih=(const float*)d_in[38], *d2_whh=(const float*)d_in[39],
              *d2_bih=(const float*)d_in[40], *d2_bhh=(const float*)d_in[41];
  const float *w4=(const float*)d_in[42], *b4=(const float*)d_in[43];
  float* out = (float*)d_out;
  float* out_zm = out + NB * NT * NIN;          // 184320
  float* out_zs = out_zm + NB * NZ;             // 200704

  char* base = (char*)d_ws;
  size_t off = 0;
  auto alloc = [&](size_t bytes) -> char* {
    off = (off + 255) & ~(size_t)255;
    char* p = base + off; off += bytes; return p;
  };
  auto abf = [&](size_t elems) -> __bf16* { return (__bf16*)alloc(elems * 2); };
  auto af  = [&](size_t elems) -> float*  { return (float*)alloc(elems * 4); };

  // packed weights (bf16, WMMA-B fragment order)
  __bf16* P_e0f_wih = abf((size_t)8192 * 96);
  __bf16* P_e0f_whh = abf((size_t)8192 * 2048);
  __bf16* P_e0b_wih = abf((size_t)8192 * 96);
  __bf16* P_e0b_whh = abf((size_t)8192 * 2048);
  __bf16* P_e1f_wih = abf((size_t)8192 * 4096);
  __bf16* P_e1f_whh = abf((size_t)8192 * 2048);
  __bf16* P_e1b_wih = abf((size_t)8192 * 4096);
  __bf16* P_e1b_whh = abf((size_t)8192 * 2048);
  __bf16* P_wm = abf((size_t)512 * 4096);
  __bf16* P_ws = abf((size_t)512 * 4096);
  __bf16* P_w2 = abf((size_t)4096 * 512);
  __bf16* P_c0_whh = abf((size_t)4096 * 1024);
  __bf16* P_c1_wih = abf((size_t)4096 * 1024);
  __bf16* P_c1_whh = abf((size_t)4096 * 1024);
  __bf16* P_w3 = abf((size_t)4096 * 1024);
  __bf16* P_d1_wih = abf((size_t)4096 * 1120);
  __bf16* P_d1_whh = abf((size_t)4096 * 1024);
  __bf16* P_d2_wih = abf((size_t)4096 * 1024);
  __bf16* P_d2_whh = abf((size_t)4096 * 1024);
  __bf16* P_w4 = abf((size_t)96 * 1024);

  // activations / states
  __bf16* xbf   = abf((size_t)NT * NB * 96);
  __bf16* y0    = abf((size_t)NT * NB * 4096);       // [t][b][0:2048 fwd | 2048:4096 bwd]
  __bf16* enc_hpp = abf((size_t)2 * NB * NH1);
  float*  enc_c   = af((size_t)NB * NH1);
  float*  hf_f32  = af((size_t)NB * NH1);
  float*  hb_f32  = af((size_t)NB * NH1);
  __bf16* ht_bf   = abf((size_t)NB * 4096);
  float*  zm_f32  = af((size_t)NB * NZ);
  float*  s_f32   = af((size_t)NB * NZ);
  __bf16* z_bf    = abf((size_t)NB * NZ);
  float*  t4_f32  = af((size_t)NB * 4096);
  __bf16* c0_hpp  = abf((size_t)2 * NB * NHC);
  float*  c0_c    = af((size_t)NB * NHC);
  __bf16* c1_hpp  = abf((size_t)2 * NB * NHC);
  float*  c1_c    = af((size_t)NB * NHC);
  __bf16* emb     = abf((size_t)NU * NB * NHC);
  __bf16* D1_hpp  = abf((size_t)2 * NB * NHD);
  float*  D1_c    = af((size_t)NB * NHD);
  __bf16* D2_hpp  = abf((size_t)2 * NB * NHD);
  float*  D2_c    = af((size_t)NB * NHD);
  __bf16* xin_d1  = abf((size_t)NB * 1120);
  float*  p_f32   = af((size_t)NB * 96);

  auto pack = [&](const float* src, __bf16* dst, int N, int K, int Np, int Kp) {
    size_t tot = (size_t)(Np / 16) * (Kp / 32) * 512;
    pack_w_kernel<<<(unsigned)((tot + 255) / 256), 256, 0, stream>>>(src, dst, N, K, Kp / 32, tot);
  };
  pack(e0f_wih, P_e0f_wih, 8192, 90, 8192, 96);
  pack(e0f_whh, P_e0f_whh, 8192, 2048, 8192, 2048);
  pack(e0b_wih, P_e0b_wih, 8192, 90, 8192, 96);
  pack(e0b_whh, P_e0b_whh, 8192, 2048, 8192, 2048);
  pack(e1f_wih, P_e1f_wih, 8192, 4096, 8192, 4096);
  pack(e1f_whh, P_e1f_whh, 8192, 2048, 8192, 2048);
  pack(e1b_wih, P_e1b_wih, 8192, 4096, 8192, 4096);
  pack(e1b_whh, P_e1b_whh, 8192, 2048, 8192, 2048);
  pack(wm, P_wm, 512, 4096, 512, 4096);
  pack(ws_, P_ws, 512, 4096, 512, 4096);
  pack(w2, P_w2, 4096, 512, 4096, 512);
  pack(c0_whh, P_c0_whh, 4096, 1024, 4096, 1024);
  pack(c1_wih, P_c1_wih, 4096, 1024, 4096, 1024);
  pack(c1_whh, P_c1_whh, 4096, 1024, 4096, 1024);
  pack(w3, P_w3, 4096, 1024, 4096, 1024);
  pack(d1_wih, P_d1_wih, 4096, 1114, 4096, 1120);
  pack(d1_whh, P_d1_whh, 4096, 1024, 4096, 1024);
  pack(d2_wih, P_d2_wih, 4096, 1024, 4096, 1024);
  pack(d2_whh, P_d2_whh, 4096, 1024, 4096, 1024);
  pack(w4, P_w4, 90, 1024, 96, 1024);

  conv_x_kernel<<<(NT * NB * 96 + 255) / 256, 256, 0, stream>>>(x, xbf);

  auto step = [&](const __bf16* xin, int Kx, const __bf16* Wih,
                  const __bf16* hin, int H, const __bf16* Whh,
                  const float* bih, const float* bhh, float* cst,
                  __bf16* hout, float* hout_f, __bf16* ysp, int ys_ld) {
    lstm_step_kernel<<<H / 64, 128, 0, stream>>>(xin, Kx, Wih, hin, H, Whh,
                                                 bih, bhh, cst, hout, hout_f, ysp, ys_ld);
  };
  auto gemm = [&](const __bf16* A, int K, const __bf16* Wp, const float* bias,
                  int N, int nbias, float* o, int act) {
    gemm_kernel<<<((N / 16) + 3) / 4, 128, 0, stream>>>(A, K, Wp, bias, N, nbias, o, act);
  };
  auto zero_enc = [&]() {
    hipMemsetAsync(enc_hpp, 0, (size_t)2 * NB * NH1 * 2, stream);
    hipMemsetAsync(enc_c, 0, (size_t)NB * NH1 * 4, stream);
  };

  // ---- encoder layer 0, forward & backward ----
  zero_enc();
  for (int t = 0; t < NT; ++t)
    step(xbf + (size_t)t * NB * 96, 96, P_e0f_wih,
         enc_hpp + (size_t)(t & 1) * NB * NH1, NH1, P_e0f_whh, e0f_bih, e0f_bhh,
         enc_c, enc_hpp + (size_t)((t + 1) & 1) * NB * NH1, nullptr,
         y0 + (size_t)t * NB * 4096, 4096);
  zero_enc();
  for (int s = 0; s < NT; ++s) {
    int t = NT - 1 - s;
    step(xbf + (size_t)t * NB * 96, 96, P_e0b_wih,
         enc_hpp + (size_t)(s & 1) * NB * NH1, NH1, P_e0b_whh, e0b_bih, e0b_bhh,
         enc_c, enc_hpp + (size_t)((s + 1) & 1) * NB * NH1, nullptr,
         y0 + (size_t)t * NB * 4096 + 2048, 4096);
  }
  // ---- encoder layer 1, forward & backward (only finals needed) ----
  zero_enc();
  for (int t = 0; t < NT; ++t)
    step(y0 + (size_t)t * NB * 4096, 4096, P_e1f_wih,
         enc_hpp + (size_t)(t & 1) * NB * NH1, NH1, P_e1f_whh, e1f_bih, e1f_bhh,
         enc_c, enc_hpp + (size_t)((t + 1) & 1) * NB * NH1, hf_f32, nullptr, 0);
  zero_enc();
  for (int s = 0; s < NT; ++s) {
    int t = NT - 1 - s;
    step(y0 + (size_t)t * NB * 4096, 4096, P_e1b_wih,
         enc_hpp + (size_t)(s & 1) * NB * NH1, NH1, P_e1b_whh, e1b_bih, e1b_bhh,
         enc_c, enc_hpp + (size_t)((s + 1) & 1) * NB * NH1, hb_f32, nullptr, 0);
  }
  // ht = concat(hf, hb) -> bf16
  f32_to_bf16_2d<<<(NB * NH1 + 255) / 256, 256, 0, stream>>>(hf_f32, ht_bf, NH1, 4096, 0, NB * NH1);
  f32_to_bf16_2d<<<(NB * NH1 + 255) / 256, 256, 0, stream>>>(hb_f32, ht_bf, NH1, 4096, 2048, NB * NH1);

  // ---- z ----
  gemm(ht_bf, 4096, P_wm, bm, NZ, NZ, zm_f32, 0);
  gemm(ht_bf, 4096, P_ws, bs, NZ, NZ, s_f32, 0);
  z_combine_kernel<<<(NB * NZ + 255) / 256, 256, 0, stream>>>(zm_f32, s_f32, eps, out_zm, out_zs, z_bf);

  // ---- conductor ----
  gemm(z_bf, NZ, P_w2, b2, 4096, 4096, t4_f32, 1);  // tanh
  init_states_kernel<<<(NB * NHC + 255) / 256, 256, 0, stream>>>(
      t4_f32, c0_hpp, c1_hpp, c0_c, c1_c, NHC);
  for (int u = 0; u < NU; ++u) {
    step(nullptr, 0, nullptr,
         c0_hpp + (size_t)(u & 1) * NB * NHC, NHC, P_c0_whh, c0_bih, c0_bhh,
         c0_c, c0_hpp + (size_t)((u + 1) & 1) * NB * NHC, nullptr, nullptr, 0);
    step(c0_hpp + (size_t)((u + 1) & 1) * NB * NHC, NHC, P_c1_wih,
         c1_hpp + (size_t)(u & 1) * NB * NHC, NHC, P_c1_whh, c1_bih, c1_bhh,
         c1_c, c1_hpp + (size_t)((u + 1) & 1) * NB * NHC, nullptr,
         emb + (size_t)u * NB * NHC, NHC);
  }

  // ---- hierarchical decoder ----
  hipMemsetAsync(p_f32, 0, (size_t)NB * 96 * 4, stream);
  for (int u = 0; u < NU; ++u) {
    const __bf16* emb_u = emb + (size_t)u * NB * NHC;
    gemm(emb_u, NHC, P_w3, b3, 4096, 4096, t4_f32, 1);  // tanh
    init_states_kernel<<<(NB * NHD + 255) / 256, 256, 0, stream>>>(
        t4_f32, D1_hpp, D2_hpp, D1_c, D2_c, NHD);
    for (int k = 0; k < NSEG; ++k) {
      build_xin_kernel<<<(NB * 1120 + 255) / 256, 256, 0, stream>>>(emb_u, p_f32, xin_d1);
      step(xin_d1, 1120, P_d1_wih,
           D1_hpp + (size_t)(k & 1) * NB * NHD, NHD, P_d1_whh, d1_bih, d1_bhh,
           D1_c, D1_hpp + (size_t)((k + 1) & 1) * NB * NHD, nullptr, nullptr, 0);
      step(D1_hpp + (size_t)((k + 1) & 1) * NB * NHD, NHD, P_d2_wih,
           D2_hpp + (size_t)(k & 1) * NB * NHD, NHD, P_d2_whh, d2_bih, d2_bhh,
           D2_c, D2_hpp + (size_t)((k + 1) & 1) * NB * NHD, nullptr, nullptr, 0);
      gemm(D2_hpp + (size_t)((k + 1) & 1) * NB * NHD, NHD, P_w4, b4, 96, 90, p_f32, 0);
      copy_p_kernel<<<(NB * NIN + 255) / 256, 256, 0, stream>>>(p_f32, out, u * NSEG + k);
    }
  }
  // ---- softmax over IN, in place ----
  softmax_kernel<<<(NB * NT + 255) / 256, 256, 0, stream>>>(out);
}
